// Faster_RCNN_5231270166937
// MI455X (gfx1250) — matrix operand order repaired and checked by reading, not compiled, not run on hardware
//
#include <hip/hip_runtime.h>

typedef _Float16 half_t;
typedef __attribute__((ext_vector_type(16))) _Float16 v16h;
typedef __attribute__((ext_vector_type(8)))  float    v8f;
typedef __attribute__((ext_vector_type(4)))  unsigned uint4v;

#define KT 32

// ------------------------------------------------------------------
// CDNA5 async global->LDS copy (ASYNCcnt-tracked), ISA §15.18 op 96.
// VDST = per-lane LDS byte offset, VADDR = 64-bit global address.
// Generic LDS pointers truncate to addr[31:0] == LDS offset (aperture rule).
// ------------------------------------------------------------------
__device__ __forceinline__ void async_ld_b32(unsigned lds_off, const void* g) {
  asm volatile("global_load_async_to_lds_b32 %0, %1, off" :: "v"(lds_off), "v"(g) : "memory");
}
__device__ __forceinline__ void wait_async0() {
  asm volatile("s_wait_asynccnt 0x0" ::: "memory");
}

// ------------------------------------------------------------------
// elementwise f32 -> f16 conversion (weights / input image)
// ------------------------------------------------------------------
__global__ void f32_to_f16_kern(const float* __restrict__ s, half_t* __restrict__ d, int n) {
  int i = blockIdx.x * blockDim.x + threadIdx.x;
  if (i < n) d[i] = (half_t)s[i];
}

// src[K][N] (f32) -> dst[N][K] (f16)  — for the FC head weights
__global__ void f32_to_f16_tr_kern(const float* __restrict__ s, half_t* __restrict__ d, int K, int N) {
  int i = blockIdx.x * blockDim.x + threadIdx.x;
  if (i < K * N) { int k = i / N, n = i - k * N; d[n * K + k] = (half_t)s[i]; }
}

// ------------------------------------------------------------------
// implicit-GEMM convolution via V_WMMA_F32_16X16X32_F16
//   D(C_out x HW) = W(C_out x Cin*KS*KS) * im2col(act)  (+BN scale/bias,
//   +optional residual, +optional ReLU).
//   64x64 tile / block, 4 wave32s, 2x2 WMMA tiles per wave.
//   LDS tiles are stored in WMMA *fragment order*:
//     tile[4 subtiles][32 lanes][8 dwords]  -> fragments load as 2x b128.
//   A (weights) staged with global_load_async_to_lds_b32 on the fast path.
// ------------------------------------------------------------------
template <int KS>
__global__ __launch_bounds__(128) void conv_wmma_kern(
    const half_t* __restrict__ act, const half_t* __restrict__ wgt,
    const float* __restrict__ scale, const float* __restrict__ bias,
    const half_t* __restrict__ res,
    half_t* __restrict__ out_h, float* __restrict__ out_f,
    int C_in, int C_out, int H_in, int W_in, int H_out, int W_out,
    int stride, int pad, int relu,
    long long ofb, long long osm, long long osp)
{
  constexpr int KSQ = KS * KS;
  __shared__ __align__(16) unsigned sAf[1024];   // A tile, fragment order
  __shared__ __align__(16) unsigned sBf[1024];   // B tile, fragment order
  __shared__ int soy[64], sox[64];               // premultiplied oy*stride, ox*stride

  const int HW  = H_out * W_out;
  const int b   = blockIdx.z;
  const int c0  = blockIdx.y * 64;
  const int p0  = blockIdx.x * 64;
  const int t    = threadIdx.x;
  const int lane = t & 31;
  const int wave = t >> 5;
  const int wm = wave >> 1, wn = wave & 1;
  const int Ktot = C_in * KSQ;
  const bool fastA = ((Ktot & 31) == 0) && (c0 + 64 <= C_out);

  // per-block pixel decomposition (runtime W_out division done 64x, not per element)
  if (t < 64) {
    int pix = p0 + t;
    if (pix < HW) {
      int oy = pix / W_out;
      soy[t] = oy * stride;
      sox[t] = (pix - oy * W_out) * stride;
    } else {
      soy[t] = -(1 << 20);   // sentinel: iy always < 0 -> zero fill
      sox[t] = 0;
    }
  }
  __syncthreads();

  v8f zero = {0.f,0.f,0.f,0.f,0.f,0.f,0.f,0.f};
  v8f acc[2][2];
  acc[0][0] = zero; acc[0][1] = zero; acc[1][0] = zero; acc[1][1] = zero;

  union H2 { half_t h[2]; unsigned u; };

  for (int k0 = 0; k0 < Ktot; k0 += KT) {
    // ---- stage A: weights 64 rows x 16 K-pairs, written in fragment order ----
    if (fastA) {
      #pragma unroll
      for (int i = 0; i < 8; ++i) {
        int idx = i * 128 + t;              // dword index 0..1023
        int m = idx >> 4, p = idx & 15;     // row, K-pair
        int kh = (p >> 2) & 1;
        int sl = (p & 3) | (((p >> 3) & 1) << 2);
        int dst = ((m >> 4) << 8) | (((m & 15) | (kh << 4)) << 3) | sl;
        unsigned lds = (unsigned)(unsigned long long)(const void*)&sAf[dst];
        async_ld_b32(lds, wgt + (long long)(c0 + m) * Ktot + (k0 + p * 2));
      }
    } else {
      #pragma unroll
      for (int i = 0; i < 8; ++i) {
        int idx = i * 128 + t;
        int m = idx >> 4, p = idx & 15;
        int kk = k0 + p * 2, mg = c0 + m;
        H2 w2; w2.h[0] = (half_t)0.f; w2.h[1] = (half_t)0.f;
        if (mg < C_out) {
          const half_t* g = wgt + (long long)mg * Ktot + kk;
          if (kk     < Ktot) w2.h[0] = g[0];
          if (kk + 1 < Ktot) w2.h[1] = g[1];
          if (kk + KT < Ktot) __builtin_prefetch(g + KT, 0, 1);
        }
        int kh = (p >> 2) & 1;
        int sl = (p & 3) | (((p >> 3) & 1) << 2);
        int dst = ((m >> 4) << 8) | (((m & 15) | (kh << 4)) << 3) | sl;
        sAf[dst] = w2.u;
      }
    }
    // ---- stage B: im2col^T 64 pixels x 16 K-pairs, fragment order ----
    #pragma unroll
    for (int i = 0; i < 8; ++i) {
      int idx = i * 128 + t;
      int n = idx & 63, p = idx >> 6;       // pixel, K-pair
      H2 v2;
      #pragma unroll
      for (int j = 0; j < 2; ++j) {
        int kk = k0 + p * 2 + j;
        half_t v = (half_t)0.f;
        if (kk < Ktot) {
          int ci = kk / KSQ;                // constant divisor -> mul/shift
          int r0 = kk - ci * KSQ;
          int iy = soy[n] + (r0 / KS) - pad;
          int ix = sox[n] + (r0 % KS) - pad;
          if (iy >= 0 && iy < H_in && ix >= 0 && ix < W_in)
            v = act[(((long long)b * C_in + ci) * H_in + iy) * W_in + ix];
        }
        v2.h[j] = v;
      }
      int dst = ((n >> 4) << 8) | (((n & 15) | (((p >> 2) & 1) << 4)) << 3) | ((p & 3) | ((p >> 3) << 2));
      sBf[dst] = v2.u;
    }
    if (fastA) wait_async0();
    __syncthreads();

    // ---- fragments: 2x b128 LDS loads each ----
    union Fr { v16h v; unsigned u[8]; };
    Fr af[2], bf[2];
    #pragma unroll
    for (int tm = 0; tm < 2; ++tm) {
      const uint4v* pa = (const uint4v*)&sAf[(((wm << 1) | tm) << 8) | (lane << 3)];
      ((uint4v*)af[tm].u)[0] = pa[0];
      ((uint4v*)af[tm].u)[1] = pa[1];
    }
    #pragma unroll
    for (int tn = 0; tn < 2; ++tn) {
      const uint4v* pb = (const uint4v*)&sBf[(((wn << 1) | tn) << 8) | (lane << 3)];
      ((uint4v*)bf[tn].u)[0] = pb[0];
      ((uint4v*)bf[tn].u)[1] = pb[1];
    }
    #pragma unroll
    for (int tm = 0; tm < 2; ++tm)
      #pragma unroll
      for (int tn = 0; tn < 2; ++tn)
        acc[tm][tn] = __builtin_amdgcn_wmma_f32_16x16x32_f16(
            false, af[tm].v, false, bf[tn].v, (short)0, acc[tm][tn], false, false);
    __syncthreads();
  }

  // ---- epilogue: BN scale/bias, residual, ReLU, f16 and/or f32 stores ----
  #pragma unroll
  for (int tm = 0; tm < 2; ++tm) {
    #pragma unroll
    for (int tn = 0; tn < 2; ++tn) {
      int pix = p0 + wn*32 + tn*16 + (lane & 15);
      if (pix >= HW) continue;
      #pragma unroll
      for (int r = 0; r < 8; ++r) {
        int mg = c0 + wm*32 + tm*16 + ((lane >> 4) << 3) + r;
        if (mg >= C_out) continue;
        float v = acc[tm][tn][r];
        if (scale) v *= scale[mg];
        if (bias)  v += bias[mg];
        long long oidx = ((long long)b * C_out + mg) * HW + pix;
        if (res)   v += (float)res[oidx];
        if (relu && v < 0.f) v = 0.f;
        if (out_h) out_h[oidx] = (half_t)v;
        if (out_f) out_f[(long long)b * ofb + (long long)mg * osm + (long long)pix * osp] = v;
      }
    }
  }
}

// ------------------------------------------------------------------
// 3x3/2 max-pool, pad 1 (stem)
// ------------------------------------------------------------------
__global__ void maxpool_kern(const half_t* __restrict__ in, half_t* __restrict__ out,
                             int H, int W, int Ho, int Wo, int total) {
  int i = blockIdx.x * blockDim.x + threadIdx.x;
  if (i >= total) return;
  int ox = i % Wo; int t2 = i / Wo; int oy = t2 % Ho; int c = t2 / Ho;  // c = b*C + c
  float m = -3.0e38f;
  int ys = oy * 2 - 1, xs = ox * 2 - 1;
  for (int dy = 0; dy < 3; ++dy) {
    int y = ys + dy; if (y < 0 || y >= H) continue;
    for (int dx = 0; dx < 3; ++dx) {
      int x = xs + dx; if (x < 0 || x >= W) continue;
      float v = (float)in[((long long)c * H + y) * W + x];
      m = fmaxf(m, v);
    }
  }
  out[i] = (half_t)m;
}

// ------------------------------------------------------------------
// per-image: sigmoid + anchor decode + validity + bitonic top-2000
// (matches lax.top_k ordering: score desc, tie -> lower index first)
// ------------------------------------------------------------------
__global__ __launch_bounds__(256) void rpn_propose_kern(
    const float* __restrict__ cls, const float* __restrict__ reg,
    float* __restrict__ boxes_all, float* __restrict__ boxes_out, int* __restrict__ keep_out)
{
  const int b = blockIdx.x;
  __shared__ float key[4096];
  __shared__ int   kid[4096];
  const float* clsb = cls + b * 2304;        // (9,16,16)
  const float* regb = reg + b * 9216;        // (36,16,16)
  float* bxall = boxes_all + (long long)b * 2304 * 4;

  for (int i = threadIdx.x; i < 4096; i += 256) { key[i] = -2.0f; kid[i] = i; }
  __syncthreads();
  for (int i = threadIdx.x; i < 2304; i += 256) {
    int a = i % 9; int pix = i / 9;
    int yy = pix >> 4, xx = pix & 15;
    float logit = clsb[a * 256 + pix];
    float score = 1.f / (1.f + expf(-logit));
    float dx = regb[(a*4+0) * 256 + pix];
    float dy = regb[(a*4+1) * 256 + pix];
    float dw = regb[(a*4+2) * 256 + pix];
    float dh = regb[(a*4+3) * 256 + pix];
    const float sizes[3]  = {128.f, 256.f, 512.f};
    const float ratios[3] = {0.5f, 1.f, 2.f};
    float s = sizes[a / 3], rr = ratios[a % 3];
    float sq = sqrtf(rr);
    float wa = s / sq, ha = s * sq;
    float cxa = (xx + 0.5f) * 32.f, cya = (yy + 0.5f) * 32.f;
    float cx = dx * wa + cxa, cy = dy * ha + cya;
    dw = fminf(fmaxf(dw, -4.f), 4.f);
    dh = fminf(fmaxf(dh, -4.f), 4.f);
    float w  = expf(dw) * wa, hh = expf(dh) * ha;
    float x1 = fminf(fmaxf(cx - 0.5f * w,  0.f), 512.f);
    float y1 = fminf(fmaxf(cy - 0.5f * hh, 0.f), 512.f);
    float x2 = fminf(fmaxf(cx + 0.5f * w,  0.f), 512.f);
    float y2 = fminf(fmaxf(cy + 0.5f * hh, 0.f), 512.f);
    bxall[i*4+0] = x1; bxall[i*4+1] = y1; bxall[i*4+2] = x2; bxall[i*4+3] = y2;
    bool valid = (x2 - x1 >= 16.f) && (y2 - y1 >= 16.f) && (score > 0.7f);
    key[i] = valid ? score : -1.0f;
  }
  __syncthreads();
  // bitonic sort, descending (ties: smaller index first)
  for (int k = 2; k <= 4096; k <<= 1) {
    for (int j = k >> 1; j > 0; j >>= 1) {
      for (int i = threadIdx.x; i < 4096; i += 256) {
        int ixj = i ^ j;
        if (ixj > i) {
          float ka = key[i], kb = key[ixj];
          int ia = kid[i], ib = kid[ixj];
          bool agb = (ka > kb) || (ka == kb && ia < ib);
          bool doswap = ((i & k) == 0) ? (!agb) : agb;
          if (doswap) { key[i] = kb; key[ixj] = ka; kid[i] = ib; kid[ixj] = ia; }
        }
      }
      __syncthreads();
    }
  }
  for (int i = threadIdx.x; i < 2000; i += 256) {
    int id = kid[i];
    float x1 = 0.f, y1 = 0.f, x2 = 0.f, y2 = 0.f;
    if (id < 2304) { x1 = bxall[id*4]; y1 = bxall[id*4+1]; x2 = bxall[id*4+2]; y2 = bxall[id*4+3]; }
    long long o = ((long long)b * 2000 + i) * 4;
    boxes_out[o+0] = x1; boxes_out[o+1] = y1; boxes_out[o+2] = x2; boxes_out[o+3] = y2;
    keep_out[b * 2000 + i] = (key[i] > 0.f) ? 1 : 0;
  }
}

// ------------------------------------------------------------------
// per-image greedy NMS (equivalent to the reference sequential scan)
// ------------------------------------------------------------------
__global__ __launch_bounds__(256) void nms_kern(const float* __restrict__ boxes, int* __restrict__ keep) {
  int b = blockIdx.x;
  __shared__ float bx[2000][4];
  __shared__ int   kp[2000];
  const float* src = boxes + (long long)b * 2000 * 4;
  for (int i = threadIdx.x; i < 2000; i += 256) {
    bx[i][0] = src[i*4]; bx[i][1] = src[i*4+1]; bx[i][2] = src[i*4+2]; bx[i][3] = src[i*4+3];
    kp[i] = keep[b * 2000 + i];
  }
  for (int i = 0; i < 2000; ++i) {
    __syncthreads();
    if (!kp[i]) continue;
    float x1 = bx[i][0], y1 = bx[i][1], x2 = bx[i][2], y2 = bx[i][3];
    float ai = (x2 - x1) * (y2 - y1);
    for (int j = i + 1 + (int)threadIdx.x; j < 2000; j += 256) {
      if (!kp[j]) continue;
      float ix1 = fmaxf(x1, bx[j][0]), iy1 = fmaxf(y1, bx[j][1]);
      float ix2 = fminf(x2, bx[j][2]), iy2 = fminf(y2, bx[j][3]);
      float inter = fmaxf(ix2 - ix1, 0.f) * fmaxf(iy2 - iy1, 0.f);
      float aj = (bx[j][2] - bx[j][0]) * (bx[j][3] - bx[j][1]);
      float iou = inter / (ai + aj - inter + 1e-6f);
      if (iou > 0.3f) kp[j] = 0;
    }
  }
  __syncthreads();
  for (int i = threadIdx.x; i < 2000; i += 256) keep[b * 2000 + i] = kp[i];
}

// ------------------------------------------------------------------
// ROI align 7x7 with 2x2 sampling -> pooled features channel-major
// pooled[c][b*2000 + r] (f16), zeroed rows for suppressed proposals
// ------------------------------------------------------------------
__global__ __launch_bounds__(256) void roi_pool_kern(
    const half_t* __restrict__ feat, const float* __restrict__ boxes,
    const int* __restrict__ keep, half_t* __restrict__ pooled)
{
  int r = blockIdx.x, b = blockIdx.y;
  int row = b * 2000 + r;
  __shared__ int   y0s[196], x0s[196], y1s[196], x1s[196];
  __shared__ float lys[196], lxs[196];
  __shared__ int sk;
  if (threadIdx.x == 0) sk = keep[row];
  float bx1 = boxes[row*4+0] * (1.f/32.f);
  float by1 = boxes[row*4+1] * (1.f/32.f);
  float bx2 = boxes[row*4+2] * (1.f/32.f);
  float by2 = boxes[row*4+3] * (1.f/32.f);
  float rw = fmaxf(bx2 - bx1, 1.f), rh = fmaxf(by2 - by1, 1.f);
  float bw = rw * (1.f/7.f), bh = rh * (1.f/7.f);
  for (int i = threadIdx.x; i < 196; i += 256) {
    int iy = i / 14, ix = i % 14;
    float offy = (float)(iy >> 1) + ((iy & 1) + 0.5f) * 0.5f;
    float offx = (float)(ix >> 1) + ((ix & 1) + 0.5f) * 0.5f;
    float y = by1 + offy * bh, x = bx1 + offx * bw;
    float y0f = floorf(y), x0f = floorf(x);
    lys[i] = y - y0f; lxs[i] = x - x0f;
    int y0 = min(max((int)y0f, 0), 15);
    int x0 = min(max((int)x0f, 0), 15);
    y0s[i] = y0; x0s[i] = x0; y1s[i] = min(y0 + 1, 15); x1s[i] = min(x0 + 1, 15);
  }
  __syncthreads();
  for (int c = threadIdx.x; c < 512; c += 256) {
    float sum = 0.f;
    if (sk) {
      const half_t* f = feat + (((long long)b * 512 + c) << 8);   // 16*16
      for (int i = 0; i < 196; ++i) {
        float a = lys[i], bl = lxs[i];
        float v00 = (float)f[(y0s[i] << 4) + x0s[i]];
        float v01 = (float)f[(y0s[i] << 4) + x1s[i]];
        float v10 = (float)f[(y1s[i] << 4) + x0s[i]];
        float v11 = (float)f[(y1s[i] << 4) + x1s[i]];
        sum += (1.f - a) * ((1.f - bl) * v00 + bl * v01) + a * ((1.f - bl) * v10 + bl * v11);
      }
      sum *= (1.f / 196.f);
    }
    pooled[(long long)c * 8000 + row] = (half_t)sum;
  }
}

// ==================================================================
// host driver
// ==================================================================
extern "C" void kernel_launch(void* const* d_in, const int* in_sizes, int n_in,
                              void* d_out, int out_size, void* d_ws, size_t ws_size,
                              hipStream_t stream) {
  (void)in_sizes; (void)n_in; (void)out_size; (void)ws_size;
  char* base = (char*)d_ws;
  size_t off = 0;
  auto alloc = [&](size_t bytes) -> void* {
    void* p = base + off; off = (off + bytes + 255) & ~(size_t)255; return p;
  };
  auto F = [&](int i) -> const float* { return (const float*)d_in[i]; };
  auto cvt = [&](int idx, long long n) -> half_t* {
    half_t* p = (half_t*)alloc((size_t)n * 2);
    f32_to_f16_kern<<<(int)((n + 255) / 256), 256, 0, stream>>>(F(idx), p, (int)n);
    return p;
  };
  auto convL = [&](const half_t* act, const half_t* w, const float* sc, const float* bi,
                   const half_t* res, half_t* oh, float* of,
                   int Cin, int Cout, int Hin, int Win, int Hout, int Wout,
                   int ks, int stride, int pad, int relu, int batch,
                   long long ofb, long long osm, long long osp) {
    dim3 g((Hout * Wout + 63) / 64, (Cout + 63) / 64, batch);
    if (ks == 1)
      conv_wmma_kern<1><<<g, 128, 0, stream>>>(act, w, sc, bi, res, oh, of,
          Cin, Cout, Hin, Win, Hout, Wout, stride, pad, relu, ofb, osm, osp);
    else if (ks == 3)
      conv_wmma_kern<3><<<g, 128, 0, stream>>>(act, w, sc, bi, res, oh, of,
          Cin, Cout, Hin, Win, Hout, Wout, stride, pad, relu, ofb, osm, osp);
    else
      conv_wmma_kern<7><<<g, 128, 0, stream>>>(act, w, sc, bi, res, oh, of,
          Cin, Cout, Hin, Win, Hout, Wout, stride, pad, relu, ofb, osm, osp);
  };

  // jax pytree (sorted-key) flatten order of setup_inputs():
  //   backbone.stages[0..3][0..1]{b1,b2,(bd),s1,s2,(sd),w1,w2,(wd)} = 0..56
  //   backbone.stem_b=57 stem_s=58 stem_w=59
  //   heads.bb=60 bc=61 wb=62 wc=63
  //   rpn.cls_b=64 cls_w=65 convs[0..3]{b,w}=66..73 reg_b=74 reg_w=75
  //   x=76
  const int IDX_X = 76;

  // ---- input image f32 -> f16 ----
  const long long NX = 4LL * 3 * 512 * 512;
  half_t* xh = (half_t*)alloc((size_t)NX * 2);
  f32_to_f16_kern<<<(int)((NX + 255) / 256), 256, 0, stream>>>(F(IDX_X), xh, (int)NX);

  // ---- stem conv 7x7/2 + BN + ReLU, then 3x3/2 maxpool ----
  half_t* stemw = cvt(59, 64LL * 3 * 49);
  half_t* stem_o = (half_t*)alloc((size_t)4 * 64 * 256 * 256 * 2);
  convL(xh, stemw, F(58), F(57), nullptr, stem_o, nullptr,
        3, 64, 512, 512, 256, 256, 7, 2, 3, 1, 4, 0, 0, 0);
  half_t* pool_o = (half_t*)alloc((size_t)4 * 64 * 128 * 128 * 2);
  int ptot = 4 * 64 * 128 * 128;
  maxpool_kern<<<(ptot + 255) / 256, 256, 0, stream>>>(stem_o, pool_o, 256, 256, 128, 128, ptot);

  // ---- backbone basic blocks ----
  struct BI { int b1, b2, bd, s1, s2, sd, w1, w2, wd; };
  const BI blk[8] = {
    { 0, 1,-1, 2, 3,-1, 4, 5,-1},
    { 6, 7,-1, 8, 9,-1,10,11,-1},
    {12,13,14,15,16,17,18,19,20},
    {21,22,-1,23,24,-1,25,26,-1},
    {27,28,29,30,31,32,33,34,35},
    {36,37,-1,38,39,-1,40,41,-1},
    {42,43,44,45,46,47,48,49,50},
    {51,52,-1,53,54,-1,55,56,-1},
  };
  int Cin = 64, H = 128;
  half_t* cur = pool_o;
  for (int bi = 0; bi < 8; ++bi) {
    const BI& K = blk[bi];
    int down = (K.wd >= 0);
    int Cout = Cin * (down ? 2 : 1);
    int stride = down ? 2 : 1;
    int Hout = H / stride;
    half_t* w1h = cvt(K.w1, (long long)Cout * Cin * 9);
    half_t* w2h = cvt(K.w2, (long long)Cout * Cout * 9);
    half_t* t1 = (half_t*)alloc((size_t)4 * Cout * Hout * Hout * 2);
    convL(cur, w1h, F(K.s1), F(K.b1), nullptr, t1, nullptr,
          Cin, Cout, H, H, Hout, Hout, 3, stride, 1, 1, 4, 0, 0, 0);
    half_t* sc = cur;
    if (down) {
      half_t* wdh = cvt(K.wd, (long long)Cout * Cin);
      sc = (half_t*)alloc((size_t)4 * Cout * Hout * Hout * 2);
      convL(cur, wdh, F(K.sd), F(K.bd), nullptr, sc, nullptr,
            Cin, Cout, H, H, Hout, Hout, 1, stride, 0, 0, 4, 0, 0, 0);
    }
    half_t* o = (half_t*)alloc((size_t)4 * Cout * Hout * Hout * 2);
    convL(t1, w2h, F(K.s2), F(K.b2), sc, o, nullptr,
          Cout, Cout, Hout, Hout, Hout, Hout, 3, 1, 1, 1, 4, 0, 0, 0);
    cur = o; Cin = Cout; H = Hout;
  }
  half_t* feat = cur;   // (4, 512, 16, 16) f16

  // ---- RPN tower: 4x (conv3x3 + bias + ReLU) ----
  const int convB[4] = {66, 68, 70, 72};
  const int convW[4] = {67, 69, 71, 73};
  half_t* h = feat; int ci = 512;
  for (int i = 0; i < 4; ++i) {
    half_t* wh = cvt(convW[i], 256LL * ci * 9);
    half_t* hn = (half_t*)alloc((size_t)4 * 256 * 256 * 2);
    convL(h, wh, nullptr, F(convB[i]), nullptr, hn, nullptr,
          ci, 256, 16, 16, 16, 16, 3, 1, 1, 1, 4, 0, 0, 0);
    h = hn; ci = 256;
  }
  // ---- RPN cls/reg 1x1 convs -> f32 NCHW maps ----
  half_t* clsw = cvt(65, 9LL * 256);
  half_t* regw = cvt(75, 36LL * 256);
  float* cls_f = (float*)alloc((size_t)4 * 9 * 256 * 4);
  float* reg_f = (float*)alloc((size_t)4 * 36 * 256 * 4);
  convL(h, clsw, nullptr, F(64), nullptr, nullptr, cls_f,
        256, 9, 16, 16, 16, 16, 1, 1, 0, 0, 4, 9LL * 256, 256, 1);
  convL(h, regw, nullptr, F(74), nullptr, nullptr, reg_f,
        256, 36, 16, 16, 16, 16, 1, 1, 0, 0, 4, 36LL * 256, 256, 1);

  // ---- proposals: decode + top-2000 + NMS + ROI pool ----
  float* boxes_all = (float*)alloc((size_t)4 * 2304 * 4 * 4);
  float* boxes     = (float*)alloc((size_t)4 * 2000 * 4 * 4);
  int*   keep      = (int*)alloc((size_t)4 * 2000 * 4);
  rpn_propose_kern<<<4, 256, 0, stream>>>(cls_f, reg_f, boxes_all, boxes, keep);
  nms_kern<<<4, 256, 0, stream>>>(boxes, keep);
  half_t* pooled = (half_t*)alloc((size_t)512 * 8000 * 2);
  roi_pool_kern<<<dim3(2000, 4), 256, 0, stream>>>(feat, boxes, keep, pooled);

  // ---- FC heads as 1x1 "conv" over 8000 ROI rows (channel-major pooled) ----
  half_t* wct = (half_t*)alloc((size_t)21 * 512 * 2);
  f32_to_f16_tr_kern<<<(512 * 21 + 255) / 256, 256, 0, stream>>>(F(63), wct, 512, 21);
  half_t* wbt = (half_t*)alloc((size_t)84 * 512 * 2);
  f32_to_f16_tr_kern<<<(512 * 84 + 255) / 256, 256, 0, stream>>>(F(62), wbt, 512, 84);
  float* outp = (float*)d_out;
  convL(pooled, wct, nullptr, F(61), nullptr, nullptr, outp,
        512, 21, 8000, 1, 8000, 1, 1, 1, 0, 0, 1, 0, 1, 21);       // (8000,21) row-major
  convL(pooled, wbt, nullptr, F(60), nullptr, nullptr, outp + 8000 * 21,
        512, 84, 8000, 1, 8000, 1, 1, 1, 0, 0, 1, 0, 1, 84);       // (8000,84) row-major
}
